// CondConvInvertedResidual_70222715289957
// MI455X (gfx1250) — matrix operand order repaired and checked by reading, not compiled, not run on hardware
//
#include <hip/hip_runtime.h>
#include <hip/hip_bf16.h>

// ---------------------------------------------------------------------------
// CondConv inverted residual for MI455X (gfx1250, wave32, WMMA bf16 path)
// B=32, C_IN=C_OUT=96, H=W=56 (HW=3136), E=8, HID=576, K=3
// ---------------------------------------------------------------------------

#define BB    32
#define CIN   96
#define COUT  96
#define HH    56
#define WW2   56
#define HW    3136
#define EE    8
#define HID   576
#define EPS   1e-5f

typedef __attribute__((ext_vector_type(16))) __bf16 v16bf;
typedef __attribute__((ext_vector_type(8)))  float  v8f;
typedef __attribute__((ext_vector_type(4)))  unsigned int v4u;
typedef __attribute__((ext_vector_type(2)))  unsigned int v2u;
typedef __attribute__((ext_vector_type(4)))  float  v4f;

union FragBF { v16bf v; v4u q[2]; };

__device__ __forceinline__ float sigmoidf_(float z) {
  return 1.0f / (1.0f + __expf(-z));
}

__device__ __forceinline__ unsigned pack_bf16(float a, float b) {
  union { __bf16 h[2]; unsigned u; } u;
  u.h[0] = (__bf16)a;
  u.h[1] = (__bf16)b;
  return u.u;
}

// ---------------------------------------------------------------------------
// 0) BN fold tables: inv = g*rsqrt(v+eps), add = beta - mean*inv, per channel
//    slots: [0..HID) BN1, [HID..2*HID) BN2, [2*HID..2*HID+COUT) BN3
// ---------------------------------------------------------------------------
__global__ __launch_bounds__(256)
void k_bnprep(const float* __restrict__ g1, const float* __restrict__ bt1,
              const float* __restrict__ m1, const float* __restrict__ v1,
              const float* __restrict__ g2, const float* __restrict__ bt2,
              const float* __restrict__ m2, const float* __restrict__ v2,
              const float* __restrict__ g3, const float* __restrict__ bt3,
              const float* __restrict__ m3, const float* __restrict__ v3,
              float* __restrict__ inv_t, float* __restrict__ add_t) {
  const int i = blockIdx.x * 256 + threadIdx.x;
  if (i < HID) {
    float inv = g1[i] * rsqrtf(v1[i] + EPS);
    inv_t[i] = inv;
    add_t[i] = bt1[i] - m1[i] * inv;
  }
  if (i < HID) {
    float inv = g2[i] * rsqrtf(v2[i] + EPS);
    inv_t[HID + i] = inv;
    add_t[HID + i] = bt2[i] - m2[i] * inv;
  }
  if (i < COUT) {
    float inv = g3[i] * rsqrtf(v3[i] + EPS);
    inv_t[2 * HID + i] = inv;
    add_t[2 * HID + i] = bt3[i] - m3[i] * inv;
  }
}

// ---------------------------------------------------------------------------
// 1) per-(b,c) mean of x over HW (vectorized float4 loads)
// ---------------------------------------------------------------------------
__global__ __launch_bounds__(256)
void k_xmean(const float* __restrict__ x, float* __restrict__ xmean) {
  const int bc = blockIdx.x;                 // b*CIN + c
  const v4f* p = (const v4f*)(x + (size_t)bc * HW);   // HW % 4 == 0
  float s = 0.f;
  for (int i = threadIdx.x; i < HW / 4; i += 256) {
    v4f v = p[i];
    s += v.x + v.y + v.z + v.w;
  }
  __shared__ float red[256];
  red[threadIdx.x] = s;
  __syncthreads();
  for (int off = 128; off > 0; off >>= 1) {
    if (threadIdx.x < off) red[threadIdx.x] += red[threadIdx.x + off];
    __syncthreads();
  }
  if (threadIdx.x == 0) xmean[bc] = red[0] * (1.0f / HW);
}

// ---------------------------------------------------------------------------
// 2) routers 1 and 3 (block router): [B,E] sigmoid(xmean @ W^T + b)
// ---------------------------------------------------------------------------
__global__ __launch_bounds__(256)
void k_router13(const float* __restrict__ xmean,
                const float* __restrict__ wr1, const float* __restrict__ br1,
                const float* __restrict__ wr3, const float* __restrict__ br3,
                float* __restrict__ r1, float* __restrict__ rblk) {
  const int t = threadIdx.x;                  // t = b*E + e, 256 = B*E
  const int b = t >> 3, e = t & 7;
  const float* xm = xmean + b * CIN;
  float a1 = br1[e], a3 = br3[e];
  for (int c = 0; c < CIN; ++c) {
    float xv = xm[c];
    a1 += xv * wr1[e * CIN + c];
    a3 += xv * wr3[e * CIN + c];
  }
  r1[t]   = sigmoidf_(a1);
  rblk[t] = sigmoidf_(a3);
}

// ---------------------------------------------------------------------------
// 3) mix expert banks for the two 1x1 convs; fold BN scale; emit bf16
// ---------------------------------------------------------------------------
__global__ __launch_bounds__(256)
void k_mix_w1(const float* __restrict__ r1, const float* __restrict__ w1,
              const float* __restrict__ inv_t, __bf16* __restrict__ w1c) {
  const int n = BB * HID * CIN;
  for (int i = blockIdx.x * 256 + threadIdx.x; i < n; i += gridDim.x * 256) {
    int c = i % CIN, o = (i / CIN) % HID, b = i / (CIN * HID);
    float acc = 0.f;
    #pragma unroll
    for (int e = 0; e < EE; ++e)
      acc += r1[b * EE + e] * w1[(e * HID + o) * CIN + c];
    w1c[i] = (__bf16)(acc * inv_t[o]);
  }
}

__global__ __launch_bounds__(256)
void k_mix_w3(const float* __restrict__ rblk, const float* __restrict__ w3,
              const float* __restrict__ inv_t, __bf16* __restrict__ w3c) {
  const int n = BB * COUT * HID;
  for (int i = blockIdx.x * 256 + threadIdx.x; i < n; i += gridDim.x * 256) {
    int c = i % HID, o = (i / HID) % COUT, b = i / (HID * COUT);
    float acc = 0.f;
    #pragma unroll
    for (int e = 0; e < EE; ++e)
      acc += rblk[b * EE + e] * w3[(e * COUT + o) * HID + c];
    w3c[i] = (__bf16)(acc * inv_t[2 * HID + o]);
  }
}

// ---------------------------------------------------------------------------
// 4) Expand GEMM (per sample): y[b] = relu6( w1c[b](HIDxCIN) * x[b](CINxHW) )
//    Tile 64(M) x 64(N); full K=96 staged in LDS once -> single barrier.
//    8 waves: 4(M) x 2(N); each wave: 16x32 strip, 2 wmma tiles, 3 K-chunks.
// ---------------------------------------------------------------------------
__global__ __launch_bounds__(256)
void k_expand_gemm(const float* __restrict__ x, const __bf16* __restrict__ w1c,
                   const float* __restrict__ add_t, __bf16* __restrict__ y) {
  const int b  = blockIdx.z;
  const int m0 = blockIdx.y * 64;
  const int n0 = blockIdx.x * 64;
  const int tid  = threadIdx.x;
  const int lane = tid & 31;
  const int wave = tid >> 5;
  const int wm = wave & 3;          // 16-row strip
  const int wn = wave >> 2;         // 32-col strip

  // row strides are multiples of 8 elements (16B) to keep b128 LDS ops aligned
  __shared__ __align__(16) __bf16 As[64][104];  // [m][k], K=96 (+8 pad)
  __shared__ __align__(16) __bf16 Bs[64][104];  // [n][k] transposed

  const float*  xb = x   + (size_t)b * CIN * HW;
  const __bf16* Ab = w1c + (size_t)b * HID * CIN;

  // ---- stage A: 64x96 bf16 = 768 x 128-bit copies, 3 per thread ----
  #pragma unroll
  for (int j = 0; j < 3; ++j) {
    const int idx = tid + j * 256;          // 0..767
    const int row = idx / 12;
    const int c8  = (idx % 12) * 8;
    *(v4u*)&As[row][c8] = *(const v4u*)&Ab[(size_t)(m0 + row) * CIN + c8];
  }
  // ---- stage B: 96(K) x 64(N) fp32 -> bf16, transposed; 6 tasks/thread ----
  #pragma unroll
  for (int j = 0; j < 6; ++j) {
    const int idx = tid + j * 256;          // 0..1535
    const int kq  = idx >> 6;               // 0..23 -> k0 = kq*4
    const int n   = idx & 63;
    const int k0  = kq * 4;
    const float* src = xb + (size_t)k0 * HW + n0 + n;
    float f0 = src[0];
    float f1 = src[HW];
    float f2 = src[2 * (size_t)HW];
    float f3 = src[3 * (size_t)HW];
    v2u p;
    p.x = pack_bf16(f0, f1);
    p.y = pack_bf16(f2, f3);
    *(v2u*)&Bs[n][k0] = p;                  // 8B aligned
  }
  __syncthreads();

  v8f c0 = {}, c1 = {};
  const int arow = wm * 16 + (lane & 15);
  const int kh   = lane >> 4;
  const int bn   = wn * 32 + (lane & 15);

  #pragma unroll
  for (int kc = 0; kc < 3; ++kc) {
    const int ko = kc * 32;
    FragBF fa, fb0, fb1;
    // A layout: lanes 0-15 -> K 0..15 of row M; lanes 16-31 -> K 16..31
    fa.q[0] = *(const v4u*)&As[arow][ko + kh * 16];
    fa.q[1] = *(const v4u*)&As[arow][ko + kh * 16 + 8];
    // B layout: lanes 0-15 -> K 0..15 of col N; lanes 16-31 -> K 16..31
    fb0.q[0] = *(const v4u*)&Bs[bn][ko + kh * 16];
    fb0.q[1] = *(const v4u*)&Bs[bn][ko + kh * 16 + 8];
    fb1.q[0] = *(const v4u*)&Bs[bn + 16][ko + kh * 16];
    fb1.q[1] = *(const v4u*)&Bs[bn + 16][ko + kh * 16 + 8];

    c0 = __builtin_amdgcn_wmma_f32_16x16x32_bf16(false, fa.v, false, fb0.v,
                                                 (short)0, c0, false, false);
    c1 = __builtin_amdgcn_wmma_f32_16x16x32_bf16(false, fa.v, false, fb1.v,
                                                 (short)0, c1, false, false);
  }

  // epilogue: +BN bias (scale folded into A), ReLU6, bf16 store
  __bf16* yb = y + (size_t)b * HID * HW;
  const int nc = n0 + wn * 32 + (lane & 15);
  #pragma unroll
  for (int r = 0; r < 8; ++r) {
    const int mrow = m0 + wm * 16 + r + ((lane >> 4) << 3);
    const float add = add_t[mrow];
    float o0 = fminf(fmaxf(c0[r] + add, 0.f), 6.f);
    float o1 = fminf(fmaxf(c1[r] + add, 0.f), 6.f);
    yb[(size_t)mrow * HW + nc]      = (__bf16)o0;
    yb[(size_t)mrow * HW + nc + 16] = (__bf16)o1;
  }
}

// ---------------------------------------------------------------------------
// 5) per-(b,hid) mean of y (8x bf16 vector loads)
// ---------------------------------------------------------------------------
__global__ __launch_bounds__(256)
void k_ymean(const __bf16* __restrict__ y, float* __restrict__ ymean) {
  const int bh = blockIdx.x;
  const __bf16* p = y + (size_t)bh * HW;
  float s = 0.f;
  for (int i = threadIdx.x; i < HW / 8; i += 256) {   // 392 vec8 chunks
    union { v4u u; __bf16 h[8]; } d;
    d.u = *(const v4u*)&p[i * 8];
    #pragma unroll
    for (int j = 0; j < 8; ++j) s += (float)d.h[j];
  }
  __shared__ float red[256];
  red[threadIdx.x] = s;
  __syncthreads();
  for (int off = 128; off > 0; off >>= 1) {
    if (threadIdx.x < off) red[threadIdx.x] += red[threadIdx.x + off];
    __syncthreads();
  }
  if (threadIdx.x == 0) ymean[bh] = red[0] * (1.0f / HW);
}

// ---------------------------------------------------------------------------
// 6) router 2
// ---------------------------------------------------------------------------
__global__ __launch_bounds__(256)
void k_router2(const float* __restrict__ ymean,
               const float* __restrict__ wr2, const float* __restrict__ br2,
               float* __restrict__ r2) {
  const int t = threadIdx.x;                  // b*E + e
  const int b = t >> 3, e = t & 7;
  float a = br2[e];
  const float* ym = ymean + b * HID;
  for (int c = 0; c < HID; ++c) a += ym[c] * wr2[e * HID + c];
  r2[t] = sigmoidf_(a);
}

// ---------------------------------------------------------------------------
// 7) mix depthwise taps, fold BN2 scale: kw[b][c][9]
// ---------------------------------------------------------------------------
__global__ __launch_bounds__(256)
void k_mix_w2(const float* __restrict__ r2, const float* __restrict__ w2,
              const float* __restrict__ inv_t, float* __restrict__ kw) {
  const int n = BB * HID;
  for (int i = blockIdx.x * 256 + threadIdx.x; i < n; i += gridDim.x * 256) {
    int c = i % HID, b = i / HID;
    const float inv = inv_t[HID + c];
    float gate[EE];
    #pragma unroll
    for (int e = 0; e < EE; ++e) gate[e] = r2[b * EE + e];
    #pragma unroll
    for (int t = 0; t < 9; ++t) {
      float acc = 0.f;
      #pragma unroll
      for (int e = 0; e < EE; ++e) acc += gate[e] * w2[(e * HID + c) * 9 + t];
      kw[(size_t)i * 9 + t] = acc * inv;
    }
  }
}

// ---------------------------------------------------------------------------
// 8) depthwise 3x3 (SAME) + BN2 + ReLU6 -> y2 bf16.
//    4 output pixels per task; 64-bit row loads + 2 halo loads per row.
// ---------------------------------------------------------------------------
__global__ __launch_bounds__(256)
void k_dwconv(const __bf16* __restrict__ y, const float* __restrict__ kw,
              const float* __restrict__ add_t, __bf16* __restrict__ y2) {
  const int bh = blockIdx.x;                    // b*HID + c
  const int c  = bh % HID;
  const __bf16* yc = y + (size_t)bh * HW;
  __bf16* oc = y2 + (size_t)bh * HW;
  float k[9];
  #pragma unroll
  for (int t = 0; t < 9; ++t) k[t] = kw[(size_t)bh * 9 + t];
  const float add = add_t[HID + c];

  const int NT = HH * (WW2 / 4);                // 784 quad tasks
  for (int t = threadIdx.x; t < NT; t += 256) {
    const int h  = t / (WW2 / 4);
    const int w0 = (t % (WW2 / 4)) * 4;
    float in[3][6];
    #pragma unroll
    for (int dy = 0; dy < 3; ++dy) {
      const int hh = h + dy - 1;
      if (hh < 0 || hh >= HH) {
        #pragma unroll
        for (int j = 0; j < 6; ++j) in[dy][j] = 0.f;
        continue;
      }
      const __bf16* rp = yc + hh * WW2 + w0;
      union { v2u u; __bf16 hx[4]; } m;
      m.u = *(const v2u*)rp;                    // 8B aligned
      in[dy][1] = (float)m.hx[0];
      in[dy][2] = (float)m.hx[1];
      in[dy][3] = (float)m.hx[2];
      in[dy][4] = (float)m.hx[3];
      in[dy][0] = (w0 > 0)         ? (float)rp[-1] : 0.f;
      in[dy][5] = (w0 + 4 < WW2)   ? (float)rp[4]  : 0.f;
    }
    union { v2u u; __bf16 hx[4]; } s;
    #pragma unroll
    for (int j = 0; j < 4; ++j) {
      float acc = 0.f;
      #pragma unroll
      for (int dy = 0; dy < 3; ++dy)
        #pragma unroll
        for (int dx = 0; dx < 3; ++dx)
          acc += k[dy * 3 + dx] * in[dy][j + dx];
      s.hx[j] = (__bf16)fminf(fmaxf(acc + add, 0.f), 6.f);
    }
    *(v2u*)(oc + h * WW2 + w0) = s.u;
  }
}

// ---------------------------------------------------------------------------
// 9) Project GEMM: out[b] = w3c[b](COUTxHID) * y2[b](HIDxHW) + bias3 + x[b]
//    Tile 64x64, K staged in 64-chunks (9 iterations).
// ---------------------------------------------------------------------------
__global__ __launch_bounds__(256)
void k_proj_gemm(const __bf16* __restrict__ y2, const __bf16* __restrict__ w3c,
                 const float* __restrict__ x, const float* __restrict__ add_t,
                 float* __restrict__ out) {
  const int b  = blockIdx.z;
  const int m0 = blockIdx.y * 64;
  const int n0 = blockIdx.x * 64;
  const int tid  = threadIdx.x;
  const int lane = tid & 31;
  const int wave = tid >> 5;
  const int wm = wave & 3;
  const int wn = wave >> 2;

  __shared__ __align__(16) __bf16 As[64][72];   // [m][k] 64-wide K (+8 pad)
  __shared__ __align__(16) __bf16 Bs[64][72];   // [n][k] transposed

  const __bf16* Ab = w3c + (size_t)b * COUT * HID;
  const __bf16* Bb = y2  + (size_t)b * HID * HW;

  v8f c0 = {}, c1 = {};
  const int arow = wm * 16 + (lane & 15);
  const int kh   = lane >> 4;
  const int bn   = wn * 32 + (lane & 15);

  for (int kk = 0; kk < HID; kk += 64) {
    // ---- stage A: 64x64 bf16 = 512 x 128-bit copies, 2 per thread ----
    #pragma unroll
    for (int j = 0; j < 2; ++j) {
      const int idx = tid + j * 256;            // 0..511
      const int row = idx >> 3;
      const int c8  = (idx & 7) * 8;
      const int mr  = m0 + row;
      v4u val = {0u, 0u, 0u, 0u};
      if (mr < COUT) val = *(const v4u*)&Ab[(size_t)mr * HID + kk + c8];
      *(v4u*)&As[row][c8] = val;
    }
    // ---- stage B: 64(K) x 64(N) bf16, transposed; 4 tasks/thread ----
    #pragma unroll
    for (int j = 0; j < 4; ++j) {
      const int idx = tid + j * 256;            // 0..1023
      const int kr  = idx >> 4;                 // 0..63
      const int n   = (idx & 15) * 4;
      const __bf16* src = &Bb[(size_t)(kk + kr) * HW + n0 + n];
      union { v2u u; __bf16 h[4]; } d;
      d.u = *(const v2u*)src;                   // 8B aligned
      Bs[n + 0][kr] = d.h[0];
      Bs[n + 1][kr] = d.h[1];
      Bs[n + 2][kr] = d.h[2];
      Bs[n + 3][kr] = d.h[3];
      if (j == 0 && kk + 64 < HID)
        __builtin_prefetch(src + (size_t)64 * HW, 0, 1);  // next K-slab
    }
    __syncthreads();

    #pragma unroll
    for (int kc = 0; kc < 2; ++kc) {
      const int ko = kc * 32;
      FragBF fa, fb0, fb1;
      fa.q[0] = *(const v4u*)&As[arow][ko + kh * 16];
      fa.q[1] = *(const v4u*)&As[arow][ko + kh * 16 + 8];
      fb0.q[0] = *(const v4u*)&Bs[bn][ko + kh * 16];
      fb0.q[1] = *(const v4u*)&Bs[bn][ko + kh * 16 + 8];
      fb1.q[0] = *(const v4u*)&Bs[bn + 16][ko + kh * 16];
      fb1.q[1] = *(const v4u*)&Bs[bn + 16][ko + kh * 16 + 8];

      c0 = __builtin_amdgcn_wmma_f32_16x16x32_bf16(false, fa.v, false, fb0.v,
                                                   (short)0, c0, false, false);
      c1 = __builtin_amdgcn_wmma_f32_16x16x32_bf16(false, fa.v, false, fb1.v,
                                                   (short)0, c1, false, false);
    }
    __syncthreads();
  }

  const int nc = n0 + wn * 32 + (lane & 15);
  #pragma unroll
  for (int r = 0; r < 8; ++r) {
    const int mrow = m0 + wm * 16 + r + ((lane >> 4) << 3);
    if (mrow >= COUT) continue;
    const float add = add_t[2 * HID + mrow];
    const size_t base = ((size_t)b * COUT + mrow) * HW;
    out[base + nc]      = c0[r] + add + x[base + nc];
    out[base + nc + 16] = c1[r] + add + x[base + nc + 16];
  }
}

// ---------------------------------------------------------------------------
// host-side launch
// ---------------------------------------------------------------------------
extern "C" void kernel_launch(void* const* d_in, const int* in_sizes, int n_in,
                              void* d_out, int out_size, void* d_ws, size_t ws_size,
                              hipStream_t stream) {
  const float* x    = (const float*)d_in[0];
  const float* wr1  = (const float*)d_in[1];
  const float* br1  = (const float*)d_in[2];
  const float* w1   = (const float*)d_in[3];
  const float* g1   = (const float*)d_in[4];
  const float* bt1  = (const float*)d_in[5];
  const float* m1   = (const float*)d_in[6];
  const float* v1   = (const float*)d_in[7];
  const float* wr2  = (const float*)d_in[8];
  const float* br2  = (const float*)d_in[9];
  const float* w2   = (const float*)d_in[10];
  const float* g2   = (const float*)d_in[11];
  const float* bt2  = (const float*)d_in[12];
  const float* m2   = (const float*)d_in[13];
  const float* v2   = (const float*)d_in[14];
  const float* w3   = (const float*)d_in[15];
  const float* g3   = (const float*)d_in[16];
  const float* bt3  = (const float*)d_in[17];
  const float* m3   = (const float*)d_in[18];
  const float* v3   = (const float*)d_in[19];
  const float* wr3  = (const float*)d_in[20];
  const float* br3  = (const float*)d_in[21];
  float* out = (float*)d_out;

  // workspace carve-out (256B aligned slices)
  char* wsp = (char*)d_ws;
  auto carve = [&](size_t bytes) -> char* {
    char* p = wsp;
    wsp += (bytes + 255) & ~(size_t)255;
    return p;
  };
  float*  xmean = (float*)carve((size_t)BB * CIN * 4);
  float*  r1    = (float*)carve((size_t)BB * EE * 4);
  float*  rblk  = (float*)carve((size_t)BB * EE * 4);
  float*  r2    = (float*)carve((size_t)BB * EE * 4);
  float*  ymean = (float*)carve((size_t)BB * HID * 4);
  float*  inv_t = (float*)carve((size_t)(2 * HID + COUT) * 4);
  float*  add_t = (float*)carve((size_t)(2 * HID + COUT) * 4);
  float*  kw    = (float*)carve((size_t)BB * HID * 9 * 4);
  __bf16* w1c   = (__bf16*)carve((size_t)BB * HID * CIN * 2);
  __bf16* w3c   = (__bf16*)carve((size_t)BB * COUT * HID * 2);
  __bf16* y     = (__bf16*)carve((size_t)BB * HID * HW * 2);
  __bf16* y2    = (__bf16*)carve((size_t)BB * HID * HW * 2);

  // 0) BN fold tables
  k_bnprep<<<(HID + 255) / 256, 256, 0, stream>>>(g1, bt1, m1, v1, g2, bt2, m2,
                                                  v2, g3, bt3, m3, v3, inv_t, add_t);

  // 1) routers on x
  k_xmean<<<BB * CIN, 256, 0, stream>>>(x, xmean);
  k_router13<<<1, 256, 0, stream>>>(xmean, wr1, br1, wr3, br3, r1, rblk);

  // 2) mixed weights (BN-scale folded), bf16
  k_mix_w1<<<(BB * HID * CIN + 255) / 256, 256, 0, stream>>>(r1, w1, inv_t, w1c);
  k_mix_w3<<<(BB * COUT * HID + 255) / 256, 256, 0, stream>>>(rblk, w3, inv_t, w3c);

  // 3) expand 1x1 GEMM (WMMA bf16) + ReLU6
  {
    dim3 grid(HW / 64, HID / 64, BB);   // 49 x 9 x 32
    k_expand_gemm<<<grid, 256, 0, stream>>>(x, w1c, add_t, y);
  }

  // 4) router2 from y means, then mixed depthwise taps
  k_ymean<<<BB * HID, 256, 0, stream>>>(y, ymean);
  k_router2<<<1, 256, 0, stream>>>(ymean, wr2, br2, r2);
  k_mix_w2<<<(BB * HID + 255) / 256, 256, 0, stream>>>(r2, w2, inv_t, kw);

  // 5) depthwise 3x3 + ReLU6
  k_dwconv<<<BB * HID, 256, 0, stream>>>(y, kw, add_t, y2);

  // 6) project 1x1 GEMM (WMMA bf16) + BN + residual
  {
    dim3 grid(HW / 64, (COUT + 63) / 64, BB);  // 49 x 2 x 32
    k_proj_gemm<<<grid, 256, 0, stream>>>(y2, w3c, x, add_t, out);
  }
}
